// GPUOptimizedMoE_50929722196049
// MI455X (gfx1250) — matrix operand order repaired and checked by reading, not compiled, not run on hardware
//
#include <hip/hip_runtime.h>
#include <hip/hip_bf16.h>
#include <math.h>

#define TOK 4096          // B*S
#define DM  768           // D
#define NE  8             // experts
#define FF  3072          // F

typedef __attribute__((ext_vector_type(16))) __bf16 bf16x16;
typedef __attribute__((ext_vector_type(8)))  float  f32x8;

union Frag16 { uint4 q[2]; bf16x16 v; };

__device__ __forceinline__ unsigned short f2bf(float f) {
  unsigned u = __float_as_uint(f);
  u += 0x7FFFu + ((u >> 16) & 1u);        // round-to-nearest-even
  return (unsigned short)(u >> 16);
}

// A fragment (16x32 bf16): two contiguous 16B segments per lane (ISA 7.12.2)
__device__ __forceinline__ bf16x16 load_afrag(const unsigned short* p0,
                                              const unsigned short* p1) {
  Frag16 a;
  a.q[0] = *(const uint4*)p0;
  a.q[1] = *(const uint4*)p1;
  return a.v;
}

// B fragment (32x16 bf16): 16 contiguous bf16 K-values per lane (32B)
__device__ __forceinline__ bf16x16 load_bfrag(const unsigned short* p) {
  Frag16 b;
  b.q[0] = *(const uint4*)p;
  b.q[1] = *(const uint4*)(p + 8);
  return b.v;
}

#define WMMA_BF16(A, B, C) \
  __builtin_amdgcn_wmma_f32_16x16x32_bf16(false, (A), false, (B), (short)0, (C), false, false)

// ---------------- one-time fp32 -> bf16 weight conversion ----------------
__global__ void __launch_bounds__(256) moe_cvt_w(
    const float* __restrict__ W1, const float* __restrict__ W2,
    unsigned short* __restrict__ W1b, unsigned short* __restrict__ W2b) {
  const size_t n1 = (size_t)NE * FF * DM;
  size_t i = ((size_t)blockIdx.x * 256 + threadIdx.x) * 8;
  const float* src; unsigned short* dst; size_t off;
  if (i < n1) { src = W1; dst = W1b; off = i; }
  else        { src = W2; dst = W2b; off = i - n1; }
  float4 a = *(const float4*)(src + off);
  float4 b = *(const float4*)(src + off + 4);
  uint4 o;
  o.x = (unsigned)f2bf(a.x) | ((unsigned)f2bf(a.y) << 16);
  o.y = (unsigned)f2bf(a.z) | ((unsigned)f2bf(a.w) << 16);
  o.z = (unsigned)f2bf(b.x) | ((unsigned)f2bf(b.y) << 16);
  o.w = (unsigned)f2bf(b.z) | ((unsigned)f2bf(b.w) << 16);
  *(uint4*)(dst + off) = o;
}

// ---------------- init ----------------
__global__ void moe_init(int* counts, int* cursor, float* gateSum) {
  int i = threadIdx.x;
  if (i < NE) { counts[i] = 0; cursor[i] = 0; }
  if (i == 0) *gateSum = 0.0f;
}

// ---------------- router (+ x -> bf16 conversion) ----------------
// cbias is constant over the expert axis -> cancels exactly in softmax/argmax.
__global__ void __launch_bounds__(256) moe_router(
    const float* __restrict__ x, const float* __restrict__ Wr,
    float* __restrict__ gates, int* __restrict__ eidx,
    int* __restrict__ counts, float* __restrict__ gateSum,
    unsigned short* __restrict__ xb) {
  __shared__ float sWr[NE * DM];                 // 24 KB
  int tid = threadIdx.x;
  for (int i = tid; i < NE * DM; i += 256) sWr[i] = Wr[i];
  __syncthreads();

  int t = blockIdx.x * 256 + tid;
  const float* xr = x + (size_t)t * DM;
  unsigned short* xbr = xb + (size_t)t * DM;

  float acc[NE];
#pragma unroll
  for (int e = 0; e < NE; ++e) acc[e] = 0.0f;

  for (int d = 0; d < DM; d += 4) {
    float4 xv = *(const float4*)(xr + d);
    unsigned lo = (unsigned)f2bf(xv.x) | ((unsigned)f2bf(xv.y) << 16);
    unsigned hi = (unsigned)f2bf(xv.z) | ((unsigned)f2bf(xv.w) << 16);
    *(uint2*)(xbr + d) = make_uint2(lo, hi);
#pragma unroll
    for (int e = 0; e < NE; ++e) {
      float4 wv = *(const float4*)&sWr[e * DM + d];
      acc[e] += xv.x * wv.x + xv.y * wv.y + xv.z * wv.z + xv.w * wv.w;
    }
  }
  float m = acc[0]; int bi = 0;
#pragma unroll
  for (int e = 1; e < NE; ++e)
    if (acc[e] > m) { m = acc[e]; bi = e; }   // strict '>' == argmax first-max
  float s = 0.0f;
#pragma unroll
  for (int e = 0; e < NE; ++e) s += expf(acc[e] - m);
  float g = 1.0f / s;                          // top-1 prob

  gates[t] = g;
  eidx[t]  = bi;
  atomicAdd(&counts[bi], 1);
  atomicAdd(gateSum, g);
}

// ---------------- prefix offsets + aux outputs ----------------
__global__ void moe_scan_aux(const int* __restrict__ counts,
                             const float* __restrict__ gateSum,
                             int* __restrict__ offsets, int* __restrict__ cursor,
                             float* __restrict__ out) {
  if (threadIdx.x != 0) return;
  int c[NE]; int off = 0, active = 0;
  float meanc = 0.0f;
#pragma unroll
  for (int e = 0; e < NE; ++e) { c[e] = counts[e]; meanc += (float)c[e]; }
  meanc /= (float)NE;
  float var = 0.0f;
#pragma unroll
  for (int e = 0; e < NE; ++e) {
    offsets[e] = off; cursor[e] = off; off += c[e];
    float d = (float)c[e] - meanc; var += d * d;
    if (c[e] > 0) active++;
  }
  offsets[NE] = off;
  const size_t base = (size_t)TOK * DM;
  out[base] = 0.01f * var / (float)NE;                     // lb_loss
  int* oi = (int*)out;
#pragma unroll
  for (int e = 0; e < NE; ++e) oi[base + 1 + e] = c[e];    // counts (int32)
  out[base + 1 + NE] = *gateSum / (float)TOK;              // average_gate
  oi[base + 2 + NE] = active;                              // active_experts
}

// ---------------- build expert-sorted permutation ----------------
__global__ void moe_perm(const int* __restrict__ eidx, int* __restrict__ cursor,
                         int* __restrict__ perm) {
  int t = blockIdx.x * 256 + threadIdx.x;
  int e = eidx[t];
  int pos = atomicAdd(&cursor[e], 1);
  perm[pos] = t;
}

// ---------------- store helpers (FULL = tile entirely in-segment) ----------
template <bool FULL>
__device__ __forceinline__ void ffn1_store(const f32x8* acc, unsigned short* Hb,
                                           int segStart, int segCount, int rbase,
                                           int ncol0, int uhalf) {
#pragma unroll
  for (int nt = 0; nt < 4; ++nt) {
    int n = ncol0 + nt * 16;
#pragma unroll
    for (int v = 0; v < 8; ++v) {
      int gr = rbase + v + (uhalf << 3);
      if (FULL || gr < segCount) {
        float val = acc[nt][v];
        float gl = 0.5f * val * (1.0f + erff(val * 0.70710678118654752f));
        Hb[(size_t)(segStart + gr) * FF + n] = f2bf(gl);
      }
    }
  }
}

template <bool FULL>
__device__ __forceinline__ void ffn2_store(const f32x8* acc, float* out,
                                           const int* s_tok, const float* s_gate,
                                           int segCount, int mtile, int rloc,
                                           int ncol0, int uhalf) {
#pragma unroll
  for (int nt = 0; nt < 4; ++nt) {
    int n = ncol0 + nt * 16;
#pragma unroll
    for (int v = 0; v < 8; ++v) {
      int lr = rloc + v + (uhalf << 3);
      if (FULL || (mtile * 64 + lr) < segCount) {
        int tok = s_tok[lr];
        out[(size_t)tok * DM + n] = s_gate[lr] * acc[nt][v];
      }
    }
  }
}

// ---------------- GEMM1: H = gelu(x_gathered @ W1e^T) -----------------------
// 8 waves = 2(M:32 rows) x 4(N:64 cols); block tile M=64 x N=256.
__global__ void __launch_bounds__(256) moe_ffn1(
    const unsigned short* __restrict__ xb, const unsigned short* __restrict__ W1b,
    const int* __restrict__ perm, const int* __restrict__ offsets,
    unsigned short* __restrict__ Hb) {
  int e = blockIdx.x >> 6;
  int mtile = blockIdx.x & 63;
  int segStart = offsets[e];
  int segCount = offsets[e + 1] - segStart;
  if (mtile * 64 >= segCount) return;
  bool full = (mtile * 64 + 64) <= segCount;

  __shared__ int s_tok[64];
  int tid = threadIdx.x;
  if (tid < 64) {
    int gidx = mtile * 64 + tid;
    s_tok[tid] = perm[segStart + (gidx < segCount ? gidx : 0)];
  }
  __syncthreads();

  int lane = tid & 31, wave = tid >> 5;
  int wm2 = wave >> 2, wn4 = wave & 3;
  int col   = lane & 15;
  int uhalf = lane >> 4;
  int khalf = uhalf << 3;    // unused (A and B both use koff)
  int koff  = uhalf << 4;    // B

  const unsigned short* xrow0 = xb + (size_t)s_tok[wm2 * 32 + col] * DM;
  const unsigned short* xrow1 = xb + (size_t)s_tok[wm2 * 32 + 16 + col] * DM;

  for (int fc = 0; fc < FF; fc += 256) {
    f32x8 acc0[4] = {}, acc1[4] = {};
    const unsigned short* wb[4];
#pragma unroll
    for (int nt = 0; nt < 4; ++nt) {
      int n = fc + wn4 * 64 + nt * 16 + col;
      wb[nt] = W1b + ((size_t)e * FF + n) * DM + koff;
    }
    for (int k = 0; k < DM; k += 32) {
      bf16x16 a0 = load_afrag(xrow0 + k + koff, xrow0 + k + koff + 8);
      bf16x16 a1 = load_afrag(xrow1 + k + koff, xrow1 + k + koff + 8);
#pragma unroll
      for (int nt = 0; nt < 4; ++nt) {
        bf16x16 b = load_bfrag(wb[nt] + k);
        acc0[nt] = WMMA_BF16(a0, b, acc0[nt]);
        acc1[nt] = WMMA_BF16(a1, b, acc1[nt]);
      }
    }
    int ncol0 = fc + wn4 * 64 + col;
    int rb0 = mtile * 64 + wm2 * 32;
    if (full) {
      ffn1_store<true>(acc0, Hb, segStart, segCount, rb0,      ncol0, uhalf);
      ffn1_store<true>(acc1, Hb, segStart, segCount, rb0 + 16, ncol0, uhalf);
    } else {
      ffn1_store<false>(acc0, Hb, segStart, segCount, rb0,      ncol0, uhalf);
      ffn1_store<false>(acc1, Hb, segStart, segCount, rb0 + 16, ncol0, uhalf);
    }
  }
}

// ---------------- GEMM2: out[tok] = gate * (H @ W2e^T), scatter -------------
__global__ void __launch_bounds__(256) moe_ffn2(
    const unsigned short* __restrict__ Hb, const unsigned short* __restrict__ W2b,
    const int* __restrict__ perm, const int* __restrict__ offsets,
    const float* __restrict__ gates, float* __restrict__ out) {
  int e = blockIdx.x >> 6;
  int mtile = blockIdx.x & 63;
  int segStart = offsets[e];
  int segCount = offsets[e + 1] - segStart;
  if (mtile * 64 >= segCount) return;
  bool full = (mtile * 64 + 64) <= segCount;

  __shared__ int s_tok[64];
  __shared__ float s_gate[64];
  int tid = threadIdx.x;
  if (tid < 64) {
    int gidx = mtile * 64 + tid;
    int tk = perm[segStart + (gidx < segCount ? gidx : 0)];
    s_tok[tid] = tk;
    s_gate[tid] = gates[tk];
  }
  __syncthreads();

  int lane = tid & 31, wave = tid >> 5;
  int wm2 = wave >> 2, wn4 = wave & 3;
  int col   = lane & 15;
  int uhalf = lane >> 4;
  int khalf = uhalf << 3;
  int koff  = uhalf << 4;

  int r0 = mtile * 64 + wm2 * 32 + col;
  int r1 = r0 + 16;
  const unsigned short* arow0 = Hb + (size_t)(segStart + (r0 < segCount ? r0 : 0)) * FF;
  const unsigned short* arow1 = Hb + (size_t)(segStart + (r1 < segCount ? r1 : 0)) * FF;
  int dbase = blockIdx.y * 256 + wn4 * 64;

  const unsigned short* wb[4];
#pragma unroll
  for (int nt = 0; nt < 4; ++nt)
    wb[nt] = W2b + ((size_t)e * DM + dbase + nt * 16 + col) * FF + koff;

  f32x8 acc0[4] = {}, acc1[4] = {};
  for (int k = 0; k < FF; k += 32) {
    bf16x16 a0 = load_afrag(arow0 + k + koff, arow0 + k + koff + 8);
    bf16x16 a1 = load_afrag(arow1 + k + koff, arow1 + k + koff + 8);
#pragma unroll
    for (int nt = 0; nt < 4; ++nt) {
      bf16x16 b = load_bfrag(wb[nt] + k);
      acc0[nt] = WMMA_BF16(a0, b, acc0[nt]);
      acc1[nt] = WMMA_BF16(a1, b, acc1[nt]);
    }
  }
  int ncol0 = dbase + col;
  int rl0 = wm2 * 32;
  if (full) {
    ffn2_store<true>(acc0, out, s_tok, s_gate, segCount, mtile, rl0,      ncol0, uhalf);
    ffn2_store<true>(acc1, out, s_tok, s_gate, segCount, mtile, rl0 + 16, ncol0, uhalf);
  } else {
    ffn2_store<false>(acc0, out, s_tok, s_gate, segCount, mtile, rl0,      ncol0, uhalf);
    ffn2_store<false>(acc1, out, s_tok, s_gate, segCount, mtile, rl0 + 16, ncol0, uhalf);
  }
}

extern "C" void kernel_launch(void* const* d_in, const int* in_sizes, int n_in,
                              void* d_out, int out_size, void* d_ws, size_t ws_size,
                              hipStream_t stream) {
  const float* x  = (const float*)d_in[0];
  const float* Wr = (const float*)d_in[2];
  const float* W1 = (const float*)d_in[5];
  const float* W2 = (const float*)d_in[6];
  // d_in[1]/d_in[3]/d_in[4]: complexity bias — constant across experts, cancels.
  (void)in_sizes; (void)n_in; (void)out_size; (void)ws_size;

  float* out = (float*)d_out;
  char* ws = (char*)d_ws;

  float* gates   = (float*)ws;                      // TOK floats
  int*   eidx    = (int*)(ws + (size_t)TOK * 4);
  int*   perm    = (int*)(ws + (size_t)TOK * 8);
  int*   counts  = (int*)(ws + (size_t)TOK * 12);
  int*   cursor  = counts + NE;
  int*   offsets = cursor + NE;
  float* gateSum = (float*)(offsets + NE + 1);

  unsigned short* xb  = (unsigned short*)(ws + 65536);       // TOK*DM bf16  (6.3 MB)
  unsigned short* Hb  = xb + (size_t)TOK * DM;               // TOK*FF bf16  (25.2 MB)
  unsigned short* W1b = Hb + (size_t)TOK * FF;               // E*F*D bf16   (37.7 MB)
  unsigned short* W2b = W1b + (size_t)NE * FF * DM;          // E*D*F bf16   (37.7 MB)

  const size_t wtot = 2 * (size_t)NE * FF * DM;              // both weight arrays
  moe_cvt_w<<<(unsigned)(wtot / (8 * 256)), 256, 0, stream>>>(W1, W2, W1b, W2b);
  moe_init<<<1, 64, 0, stream>>>(counts, cursor, gateSum);
  moe_router<<<TOK / 256, 256, 0, stream>>>(x, Wr, gates, eidx, counts, gateSum, xb);
  moe_scan_aux<<<1, 32, 0, stream>>>(counts, gateSum, offsets, cursor, out);
  moe_perm<<<TOK / 256, 256, 0, stream>>>(eidx, cursor, perm);
  moe_ffn1<<<NE * (TOK / 64), 256, 0, stream>>>(xb, W1b, perm, offsets, Hb);
  moe_ffn2<<<dim3(NE * (TOK / 64), DM / 256), 256, 0, stream>>>(Hb, W2b, perm, offsets, gates, out);
}
